// OctreeConvBn_24146306138634
// MI455X (gfx1250) — matrix operand order repaired and checked by reading, not compiled, not run on hardware
//
#include <hip/hip_runtime.h>
#include <hip/hip_bf16.h>

typedef _Float16 v16h __attribute__((ext_vector_type(16)));
typedef _Float16 v8h  __attribute__((ext_vector_type(8)));
typedef float    v8f  __attribute__((ext_vector_type(8)));

#define BN_EPS 1e-3f

// ---------------------------------------------------------------------------
// Zero the 64-float stats buffer (sum[32], sumsq[32]) every launch:
// the harness poisons d_ws once and never re-zeros between replays.
// ---------------------------------------------------------------------------
__global__ void zero_stats_kernel(float* stats) {
    int t = threadIdx.x;
    if (t < 64) stats[t] = 0.0f;
}

// ---------------------------------------------------------------------------
// data_in is [1, Cin=32, N, 1] (channel-major). Convert to node-major f16
// xh[N][32] so the per-node gather in the conv kernel is one contiguous
// 64-byte row. Reads are coalesced across lanes for each c; writes are
// 64B-contiguous per lane (fully coalesced as a wave).
// ---------------------------------------------------------------------------
__global__ void prep_x_kernel(const float* __restrict__ din,
                              _Float16* __restrict__ xh, int N) {
    int n = blockIdx.x * blockDim.x + threadIdx.x;
    if (n >= N) return;
    _Float16 row[32];
#pragma unroll
    for (int c = 0; c < 32; ++c) row[c] = (_Float16)din[c * N + n];
    v8h* dst = (v8h*)(xh + (size_t)n * 32);
    const v8h* src = (const v8h*)row;
#pragma unroll
    for (int i = 0; i < 4; ++i) dst[i] = src[i];
}

// ---------------------------------------------------------------------------
// Pre-swizzle weight [27, Cin=32, Cout=32] (f32) into WMMA B fragments (f16).
// v_wmma_f32_16x16x32_f16 B layout (32x16, K x N), wave32:
//   lane L (0..15):  column N=L,    K = 0..7 then 16..23 packed in 8 VGPRs
//   lane L (16..31): column N=L-16, K = 8..15 then 24..31
// Stored flat as bfrag[((k*2 + half)*32 + lane)*16 + i], i = K offset 0..15,
// so the conv kernel loads one v16h per lane.
// ---------------------------------------------------------------------------
__global__ void prep_b_kernel(const float* __restrict__ w,
                              _Float16* __restrict__ bfrag) {
    int tid = blockIdx.x * blockDim.x + threadIdx.x;
    if (tid >= 27 * 2 * 32 * 16) return;
    int i    = tid & 15;          // K offset within lane-half
    int lane = (tid >> 4) & 31;
    int h    = (tid >> 9) & 1;    // Cout half (0: ch 0-15, 1: ch 16-31)
    int k    = tid >> 10;         // tap
    int kk = ((lane >> 4) << 3) + (i & 7) + ((i >> 3) << 4);     // Cin index 0..31
    int d  = (h << 4) + (lane & 15);     // Cout index 0..31
    bfrag[tid] = (_Float16)w[k * 1024 + kk * 32 + d];
}

// ---------------------------------------------------------------------------
// Gathered GEMM via WMMA. Each wave owns 4 M-tiles of 16 nodes (64 nodes),
// reusing each B fragment across the 4 tiles. 27 taps * 2 Cout halves
// * 4 tiles = 216 v_wmma_f32_16x16x32_f16 per wave.
//
// Fast path: waves whose 4 tiles are all in range run a completely
// branchless k-loop (no exec-mask traffic, loads cluster into clauses,
// 8 independent WMMA accumulation chains). Only the single partial tail
// wave takes the guarded slow path.
//
// A-fragment layout (16x32 f16, M x K), wave32:
//   lanes 0-15:  row M=lane,    halves = Cin 0..7  then 16..23
//   lanes 16-31: row M=lane-16, halves = Cin 8..15 then 24..31
// -> per lane: two 16-byte loads from the gathered node row at c0 and c0+16.
//
// C/D layout: lane L holds channel N=L%16; rows M = (L<16 ? 0..7 : 8..15)
// -> per-lane partial sum / sum-of-squares is for exactly one channel.
// ---------------------------------------------------------------------------
__global__ void __launch_bounds__(128)
octconv_wmma_kernel(const int* __restrict__ neigh,
                    const _Float16* __restrict__ xh,
                    const _Float16* __restrict__ bfrag,
                    float* __restrict__ y,
                    float* __restrict__ stats,
                    int N) {
    __shared__ float lsum[32];
    __shared__ float lsq[32];
    const int tid = threadIdx.x;
    if (tid < 32) { lsum[tid] = 0.0f; lsq[tid] = 0.0f; }
    __syncthreads();

    const int lane = tid & 31;
    const int wave = tid >> 5;
    const int r    = lane & 15;   // row within tile / channel within half
    const int hi   = lane >> 4;   // lane half
    const int base = (blockIdx.x * 4 + wave) * 64;
    const int c0   = hi * 8;      // Cin start for A fragment (0 or 8)

    v8f zero = {};
    v8f acc[4][2];
#pragma unroll
    for (int t = 0; t < 4; ++t) { acc[t][0] = zero; acc[t][1] = zero; }

    // Per-tile neighbor-row base pointers (row = base + t*16 + r).
    const int* np0 = neigh + (size_t)(base + 0 * 16 + r) * 27;
    const int* np1 = neigh + (size_t)(base + 1 * 16 + r) * 27;
    const int* np2 = neigh + (size_t)(base + 2 * 16 + r) * 27;
    const int* np3 = neigh + (size_t)(base + 3 * 16 + r) * 27;
    const _Float16* bp = bfrag + (size_t)lane * 16;

    if (base + 64 <= N) {
        // -------- fast path: all 4 tiles valid, fully branchless --------
#pragma unroll 1
        for (int k = 0; k < 27; ++k) {
            // Gather indices for the 4 tiles (independent loads -> clause).
            const int n0 = np0[k];
            const int n1 = np1[k];
            const int n2 = np2[k];
            const int n3 = np3[k];
            // B fragments for both Cout halves.
            const v16h b0 = *(const v16h*)(bp + (size_t)(k * 2 + 0) * 512);
            const v16h b1 = *(const v16h*)(bp + (size_t)(k * 2 + 1) * 512);
            // Gathered A rows: two 16B loads per lane per tile.
            const _Float16* x0 = xh + (size_t)n0 * 32 + c0;
            const _Float16* x1 = xh + (size_t)n1 * 32 + c0;
            const _Float16* x2 = xh + (size_t)n2 * 32 + c0;
            const _Float16* x3 = xh + (size_t)n3 * 32 + c0;
            v8h a0l = *(const v8h*)(x0), a0h = *(const v8h*)(x0 + 16);
            v8h a1l = *(const v8h*)(x1), a1h = *(const v8h*)(x1 + 16);
            v8h a2l = *(const v8h*)(x2), a2h = *(const v8h*)(x2 + 16);
            v8h a3l = *(const v8h*)(x3), a3h = *(const v8h*)(x3 + 16);
            v16h a0 = __builtin_shufflevector(a0l, a0h, 0,1,2,3,4,5,6,7,8,9,10,11,12,13,14,15);
            v16h a1 = __builtin_shufflevector(a1l, a1h, 0,1,2,3,4,5,6,7,8,9,10,11,12,13,14,15);
            v16h a2 = __builtin_shufflevector(a2l, a2h, 0,1,2,3,4,5,6,7,8,9,10,11,12,13,14,15);
            v16h a3 = __builtin_shufflevector(a3l, a3h, 0,1,2,3,4,5,6,7,8,9,10,11,12,13,14,15);
            acc[0][0] = __builtin_amdgcn_wmma_f32_16x16x32_f16(false, a0, false, b0, (short)0, acc[0][0], false, false);
            acc[0][1] = __builtin_amdgcn_wmma_f32_16x16x32_f16(false, a0, false, b1, (short)0, acc[0][1], false, false);
            acc[1][0] = __builtin_amdgcn_wmma_f32_16x16x32_f16(false, a1, false, b0, (short)0, acc[1][0], false, false);
            acc[1][1] = __builtin_amdgcn_wmma_f32_16x16x32_f16(false, a1, false, b1, (short)0, acc[1][1], false, false);
            acc[2][0] = __builtin_amdgcn_wmma_f32_16x16x32_f16(false, a2, false, b0, (short)0, acc[2][0], false, false);
            acc[2][1] = __builtin_amdgcn_wmma_f32_16x16x32_f16(false, a2, false, b1, (short)0, acc[2][1], false, false);
            acc[3][0] = __builtin_amdgcn_wmma_f32_16x16x32_f16(false, a3, false, b0, (short)0, acc[3][0], false, false);
            acc[3][1] = __builtin_amdgcn_wmma_f32_16x16x32_f16(false, a3, false, b1, (short)0, acc[3][1], false, false);
        }
    } else if (base < N) {
        // -------- tail path: per-tile guards (wave-uniform; N % 16 == 0) ----
#pragma unroll 1
        for (int k = 0; k < 27; ++k) {
            const v16h b0 = *(const v16h*)(bp + (size_t)(k * 2 + 0) * 512);
            const v16h b1 = *(const v16h*)(bp + (size_t)(k * 2 + 1) * 512);
#pragma unroll
            for (int t = 0; t < 4; ++t) {
                const int nb = base + t * 16;
                if (nb < N) {
                    const int node = neigh[(size_t)(nb + r) * 27 + k];
                    const _Float16* xr = xh + (size_t)node * 32 + c0;
                    v8h alo = *(const v8h*)(xr);
                    v8h ahi = *(const v8h*)(xr + 16);
                    v16h a = __builtin_shufflevector(alo, ahi,
                              0,1,2,3,4,5,6,7,8,9,10,11,12,13,14,15);
                    acc[t][0] = __builtin_amdgcn_wmma_f32_16x16x32_f16(
                        false, a, false, b0, (short)0, acc[t][0], false, false);
                    acc[t][1] = __builtin_amdgcn_wmma_f32_16x16x32_f16(
                        false, a, false, b1, (short)0, acc[t][1], false, false);
                }
            }
        }
    }

    // Write y[N][32] (node-major) and accumulate BN statistics.
#pragma unroll
    for (int t = 0; t < 4; ++t) {
        const int nb = base + t * 16;
        if (nb < N) {
#pragma unroll
            for (int h = 0; h < 2; ++h) {
                const int ch = h * 16 + r;
                float s = 0.0f, q = 0.0f;
#pragma unroll
                for (int v = 0; v < 8; ++v) {
                    float val = acc[t][h][v];
                    y[(size_t)(nb + hi * 8 + v) * 32 + ch] = val;
                    s += val;
                    q += val * val;
                }
                atomicAdd(&lsum[ch], s);
                atomicAdd(&lsq[ch], q);
            }
        }
    }
    __syncthreads();
    if (tid < 32) {
        atomicAdd(&stats[tid],      lsum[tid]);
        atomicAdd(&stats[32 + tid], lsq[tid]);
    }
}

// ---------------------------------------------------------------------------
// Finalize BN and write output in [1, Cout=32, N, 1] (channel-major) layout.
// ---------------------------------------------------------------------------
__global__ void bn_out_kernel(const float* __restrict__ y,
                              const float* __restrict__ stats,
                              const float* __restrict__ gamma,
                              const float* __restrict__ beta,
                              float* __restrict__ out, int N) {
    __shared__ float sm[32], si[32], sg[32], sb[32];
    int tid = threadIdx.x;
    if (tid < 32) {
        float invN = 1.0f / (float)N;
        float m = stats[tid] * invN;
        float v = stats[32 + tid] * invN - m * m;
        sm[tid] = m;
        si[tid] = rsqrtf(v + BN_EPS);
        sg[tid] = gamma[tid];
        sb[tid] = beta[tid];
    }
    __syncthreads();
    int n = blockIdx.x * blockDim.x + tid;
    if (n >= N) return;
    const float4* yr = (const float4*)(y + (size_t)n * 32);
#pragma unroll
    for (int i = 0; i < 8; ++i) {
        float4 vv = yr[i];
        int c = i * 4;
        out[(size_t)(c + 0) * N + n] = (vv.x - sm[c + 0]) * si[c + 0] * sg[c + 0] + sb[c + 0];
        out[(size_t)(c + 1) * N + n] = (vv.y - sm[c + 1]) * si[c + 1] * sg[c + 1] + sb[c + 1];
        out[(size_t)(c + 2) * N + n] = (vv.z - sm[c + 2]) * si[c + 2] * sg[c + 2] + sb[c + 2];
        out[(size_t)(c + 3) * N + n] = (vv.w - sm[c + 3]) * si[c + 3] * sg[c + 3] + sb[c + 3];
    }
}

// ---------------------------------------------------------------------------
// Workspace layout (bytes):
//   [0,      256)                 stats: sum[32], sumsq[32]
//   [256,    55552)               bfrag: 27*2*32*16 f16  (55296 B)
//   [55552,  55552 + N*64)        xh:    N*32 f16
//   [.. ,    .. + N*128)          y:     N*32 f32
// Total ~19.3 MB for N = 100000.
// ---------------------------------------------------------------------------
extern "C" void kernel_launch(void* const* d_in, const int* in_sizes, int n_in,
                              void* d_out, int out_size, void* d_ws, size_t ws_size,
                              hipStream_t stream) {
    const float* data_in = (const float*)d_in[0];   // [1,32,N,1] f32
    const int*   neigh   = (const int*)d_in[1];     // [N,27] i32
    const float* weight  = (const float*)d_in[2];   // [27,32,32] f32
    const float* gamma   = (const float*)d_in[3];   // [32] f32
    const float* beta    = (const float*)d_in[4];   // [32] f32
    float* out = (float*)d_out;

    const int N = in_sizes[1] / 27;

    char* ws = (char*)d_ws;
    float*    stats = (float*)ws;
    _Float16* bfrag = (_Float16*)(ws + 256);
    _Float16* xh    = (_Float16*)(ws + 55552);
    float*    y     = (float*)(ws + 55552 + (size_t)N * 64);

    zero_stats_kernel<<<1, 64, 0, stream>>>(stats);
    prep_x_kernel<<<(N + 255) / 256, 256, 0, stream>>>(data_in, xh, N);
    prep_b_kernel<<<(27 * 2 * 32 * 16 + 255) / 256, 256, 0, stream>>>(weight, bfrag);

    int tiles  = (N + 15) / 16;       // 16-node M-tiles
    int waves  = (tiles + 3) / 4;     // 4 tiles per wave
    int blocks = (waves + 3) / 4;     // 4 waves (128 threads) per block
    octconv_wmma_kernel<<<blocks, 128, 0, stream>>>(neigh, xh, bfrag, y, stats, N);

    bn_out_kernel<<<(N + 255) / 256, 256, 0, stream>>>(y, stats, gamma, beta, out, N);
}
